// TrackACT_55155970015684
// MI455X (gfx1250) — matrix operand order repaired and checked by reading, not compiled, not run on hardware
//
#include <hip/hip_runtime.h>
#include <hip/hip_bf16.h>

// ---------------------------------------------------------------------------
// MoE top-1 (eval, capacity never binds) for MI455X / gfx1250, wave32 + WMMA.
//   per token: e*,g = top2-gate(softmax(x.Wg)); out = g * W2[e*]^T gelu(W1[e*]^T LN(x))
// f16 WMMA 16x16x32 with fp32 accumulation; weights pre-packed to fragment
// order; B fragments double-buffered in registers; next chunk prefetched.
// ---------------------------------------------------------------------------

typedef __attribute__((ext_vector_type(16))) _Float16    v16h;
typedef __attribute__((ext_vector_type(8)))  float       v8f;
typedef __attribute__((ext_vector_type(4)))  unsigned    v4u;
typedef __attribute__((ext_vector_type(4)))  float       v4f;

#define NBATCH 4
#define SEQ    1024
#define DMODEL 512
#define NEXP   6
#define HDIM   2048
#define NTOK   (NBATCH * SEQ)   // 4096
#define MROWS  32
#define NTILES (NTOK / MROWS)   // 128

union AF { v16h h; v4u q[2]; };

// A-fragment (16x32 f16, MxK) from an LDS row-major matrix.
// ISA layout: lanes 0-15 row M=lane, K = {k0..k0+7} U {k0+16..k0+23};
//             lanes 16-31 row M=lane-16, K shifted by +8 in each run.
static __device__ inline v16h ld_afrag_lds(const _Float16* base, int rowStride,
                                           int row, int k0, int half8) {
  AF f;
  const _Float16* rp = base + row * rowStride + k0 + half8;
  f.q[0] = *(const v4u*)(rp);
  f.q[1] = *(const v4u*)(rp + 16);
  return f.h;
}

// B-fragment (32x16 f16, KxN) from pre-packed global memory: lane's 16 halves
// are contiguous (32B) -> two coalesced b128 loads per fragment.
static __device__ inline v16h ld_bfrag(const _Float16* frags, int fid, int lane) {
  AF f;
  const v4u* p = (const v4u*)(frags + (size_t)fid * 512 + lane * 16);
  f.q[0] = p[0];
  f.q[1] = p[1];
  return f.h;
}

// ---------------------------------------------------------------------------
// Pack fp32 weight matrix [NEXP][KTOT][NTOT] into f16 WMMA-B fragment order.
// Fragment (e, nb, kb) holds K = kb*32.., N = nb*16..; element (lane, j) =
// W[K = 8*(lane>>4) + j + (j>=8 ? 8 : 0)][N = lane&15].
// ---------------------------------------------------------------------------
__global__ void pack_w(const float* __restrict__ src, _Float16* __restrict__ dst,
                       int KTOT, int NTOT, int KBLK, int NBLK) {
  int gt = blockIdx.x * blockDim.x + threadIdx.x;
  int fid = gt >> 5, lane = gt & 31;
  if (fid >= NEXP * NBLK * KBLK) return;
  int e = fid / (NBLK * KBLK);
  int rem = fid - e * (NBLK * KBLK);
  int nb = rem / KBLK, kb = rem - nb * KBLK;
  int K0 = kb * 32 + 8 * (lane >> 4);
  int N0 = nb * 16 + (lane & 15);
  const float* s = src + (size_t)e * KTOT * NTOT;
  AF f;
#pragma unroll
  for (int j = 0; j < 16; ++j)
    f.h[j] = (_Float16)s[(size_t)(K0 + j + (j >= 8 ? 8 : 0)) * NTOT + N0];
  v4u* d = (v4u*)(dst + (size_t)fid * 512 + lane * 16);
  d[0] = f.q[0];
  d[1] = f.q[1];
}

// ---------------------------------------------------------------------------
// Gating + LayerNorm. One wave (32 lanes) per token; lane owns 16 contiguous d.
// ---------------------------------------------------------------------------
__global__ void gate_ln(const float* __restrict__ x, const float* __restrict__ wg,
                        const float* __restrict__ gamma, _Float16* __restrict__ xnf,
                        int* __restrict__ counts, int* __restrict__ lists,
                        float* __restrict__ gatev) {
  __shared__ float swg[DMODEL * NEXP];
  __shared__ float sgam[DMODEL];
  int tid = threadIdx.x;
  for (int i = tid; i < DMODEL * NEXP; i += blockDim.x) swg[i] = wg[i];
  for (int i = tid; i < DMODEL; i += blockDim.x) sgam[i] = gamma[i];
  __syncthreads();

  int t = blockIdx.x * 8 + (tid >> 5);
  int lane = tid & 31;
  int d0 = lane * 16;

  float xv[16];
  const v4f* xp = (const v4f*)(x + (size_t)t * DMODEL + d0);
#pragma unroll
  for (int i = 0; i < 4; ++i) {
    v4f v = xp[i];
    xv[4 * i + 0] = v[0]; xv[4 * i + 1] = v[1];
    xv[4 * i + 2] = v[2]; xv[4 * i + 3] = v[3];
  }

  float sum = 0.f, sq = 0.f, lg[NEXP];
#pragma unroll
  for (int e = 0; e < NEXP; ++e) lg[e] = 0.f;
#pragma unroll
  for (int i = 0; i < 16; ++i) {
    float v = xv[i];
    sum += v; sq += v * v;
    int d = d0 + i;
#pragma unroll
    for (int e = 0; e < NEXP; ++e) lg[e] += v * swg[d * NEXP + e];
  }
  // wave32 xor-reduction: every lane ends with the full sums
#pragma unroll
  for (int off = 16; off >= 1; off >>= 1) {
    sum += __shfl_xor(sum, off, 32);
    sq  += __shfl_xor(sq,  off, 32);
#pragma unroll
    for (int e = 0; e < NEXP; ++e) lg[e] += __shfl_xor(lg[e], off, 32);
  }
  float mu = sum * (1.f / DMODEL);
  float var = sq * (1.f / DMODEL) - mu * mu;
  float rstd = rsqrtf(var + 1e-5f);

  // softmax over 6 logits + top-2 gate (redundant per lane; lane 0 commits)
  float mx = lg[0];
#pragma unroll
  for (int e = 1; e < NEXP; ++e) mx = fmaxf(mx, lg[e]);
  float S = 0.f, p[NEXP];
#pragma unroll
  for (int e = 0; e < NEXP; ++e) { p[e] = expf(lg[e] - mx); S += p[e]; }
  float inv = 1.f / S;
  float g1 = -1.f, g2 = -1.f; int i1 = 0;
#pragma unroll
  for (int e = 0; e < NEXP; ++e) {
    float pe = p[e] * inv;
    if (pe > g1) { g2 = g1; g1 = pe; i1 = e; }
    else if (pe > g2) g2 = pe;
  }
  float g = g1 / (g1 + g2 + 1e-9f);
  if (lane == 0) {
    int slot = atomicAdd(&counts[i1], 1);
    lists[i1 * NTOK + slot] = t;
    gatev[i1 * NTOK + slot] = g;
  }

  AF f;
#pragma unroll
  for (int i = 0; i < 16; ++i)
    f.h[i] = (_Float16)((xv[i] - mu) * rstd * sgam[d0 + i]);
  v4u* dp = (v4u*)(xnf + (size_t)t * DMODEL + d0);
  dp[0] = f.q[0];
  dp[1] = f.q[1];
}

// ---------------------------------------------------------------------------
// Expert MLP: grid (tile, expert), 256 threads = 8 waves, 32 tokens per tile.
// Fused over hidden in 64-wide chunks; stage1 gelu(A@W1) -> LDS, stage2
// Out += Hc@W2 in fp32 accumulators. 32 WMMA / chunk / wave; B fragments are
// double-buffered in registers so WMMA overlaps the b128 loads.
// ---------------------------------------------------------------------------
__global__ void __launch_bounds__(256) moe_mlp(
    const _Float16* __restrict__ xnf, const _Float16* __restrict__ w1f,
    const _Float16* __restrict__ w2f, const int* __restrict__ counts,
    const int* __restrict__ lists, const float* __restrict__ gatev,
    float* __restrict__ out) {
  __shared__ alignas(16) _Float16 sA[MROWS][DMODEL];  // 32 KB, LN'ed tokens
  __shared__ alignas(16) _Float16 sH[MROWS][64];      //  4 KB, hidden chunk
  __shared__ float sG[MROWS];
  __shared__ int   sT[MROWS];

  int e = blockIdx.y;
  int cnt = counts[e];
  if (cnt > NTOK) cnt = NTOK;
  int tile0 = blockIdx.x * MROWS;
  if (tile0 >= cnt) return;   // block-uniform: WMMA below runs with full EXEC
  int tid = threadIdx.x;

  { // gather 32 token rows (f16) into LDS; pad with zeros
    int row = tid >> 3, seg = (tid & 7) * 64;
    bool valid = (tile0 + row) < cnt;
    int tok = valid ? lists[e * NTOK + tile0 + row] : -1;
    if ((tid & 7) == 0) {
      sT[row] = tok;
      sG[row] = valid ? gatev[e * NTOK + tile0 + row] : 0.f;
    }
    const v4u* src = (const v4u*)(xnf + (size_t)(valid ? tok : 0) * DMODEL + seg);
    v4u* dst = (v4u*)(&sA[row][seg]);
    v4u z = {0u, 0u, 0u, 0u};
#pragma unroll
    for (int i = 0; i < 8; ++i) dst[i] = valid ? src[i] : z;
  }
  __syncthreads();

  int w = tid >> 5, lane = tid & 31;
  int half8 = (lane >> 4) * 8, lr = lane & 15;
  int m1 = w >> 2, hs = w & 3;  // stage-1 Hc tile owned by this wave

  v8f oacc[2][4];
#pragma unroll
  for (int m = 0; m < 2; ++m)
#pragma unroll
    for (int nt = 0; nt < 4; ++nt)
#pragma unroll
      for (int i = 0; i < 8; ++i) oacc[m][nt][i] = 0.f;

#pragma unroll 1
  for (int c = 0; c < HDIM / 64; ++c) {
    int fbase1 = (e * (HDIM / 16) + c * 4 + hs) * (DMODEL / 32);
    int fbase2 = (e * (DMODEL / 16) + w * 4) * (HDIM / 32) + c * 2;

    // Warm WGP$ for the NEXT chunk's weight fragments (global_prefetch_b8).
    // Stage-1 region: 16 frags * 1KB = 16KB; stage-2 region: 8 frags spread
    // over 4 n-blocks. One prefetch per lane sweeps stage 1; in stage 2 it covers only the first n-block's 2 frags.
    if (c + 1 < HDIM / 64) {
      const _Float16* p1 = w1f + (size_t)(fbase1 + (DMODEL / 32) * 4) * 512;
      __builtin_prefetch(p1 + lane * 256, 0, 2);
      const _Float16* p2 = w2f + (size_t)(fbase2 + 2) * 512;
      __builtin_prefetch(p2 + lane * 32, 0, 2);
    }

    // ---- stage 1: Hc(16x16) = gelu( A(16x512) @ W1[:, 16 cols] ) ----
    v8f hacc;
#pragma unroll
    for (int i = 0; i < 8; ++i) hacc[i] = 0.f;

    v16h bcur = ld_bfrag(w1f, fbase1, lane);
#pragma unroll
    for (int kb = 0; kb < DMODEL / 32; ++kb) {
      v16h bnxt;
      if (kb + 1 < DMODEL / 32) bnxt = ld_bfrag(w1f, fbase1 + kb + 1, lane);
      v16h a = ld_afrag_lds(&sA[0][0], DMODEL, m1 * 16 + lr, kb * 32, half8);
      hacc = __builtin_amdgcn_wmma_f32_16x16x32_f16(false, a, false, bcur,
                                                    (short)0, hacc, false, false);
      if (kb + 1 < DMODEL / 32) bcur = bnxt;
    }
#pragma unroll
    for (int p = 0; p < 8; ++p) {   // exact GELU, write C-layout tile to LDS
      float v = hacc[p];
      float ge = 0.5f * v * (1.0f + erff(v * 0.70710678118f));
      sH[m1 * 16 + p + half8][hs * 16 + lr] = (_Float16)ge;
    }
    __syncthreads();

    // ---- stage 2: Out(32 x my-64-cols) += Hc(32x64) @ W2-chunk ----
    v16h hf[2][2];
#pragma unroll
    for (int m = 0; m < 2; ++m)
#pragma unroll
      for (int kb2 = 0; kb2 < 2; ++kb2)
        hf[m][kb2] = ld_afrag_lds(&sH[0][0], 64, m * 16 + lr, kb2 * 32, half8);

    v16h b2cur = ld_bfrag(w2f, fbase2, lane);
#pragma unroll
    for (int i = 0; i < 8; ++i) {       // i = nt*2 + kb2, double-buffered
      int nt = i >> 1, kb2 = i & 1;
      v16h b2nxt;
      if (i + 1 < 8) {
        int j = i + 1;
        b2nxt = ld_bfrag(w2f, fbase2 + (j >> 1) * (HDIM / 32) + (j & 1), lane);
      }
#pragma unroll
      for (int m = 0; m < 2; ++m)
        oacc[m][nt] = __builtin_amdgcn_wmma_f32_16x16x32_f16(
            false, hf[m][kb2], false, b2cur, (short)0, oacc[m][nt], false, false);
      if (i + 1 < 8) b2cur = b2nxt;
    }
    __syncthreads();
  }

  // ---- scale by gate and scatter to token rows ----
#pragma unroll
  for (int m = 0; m < 2; ++m) {
#pragma unroll
    for (int p = 0; p < 8; ++p) {
      int r = m * 16 + p + half8;
      int t = sT[r];
      if (t < 0) continue;
      float g = sG[r];
      float* op = out + (size_t)t * DMODEL + w * 64 + lr;
#pragma unroll
      for (int nt = 0; nt < 4; ++nt)
        op[nt * 16] = g * oacc[m][nt][p];
    }
  }
}

// ---------------------------------------------------------------------------
extern "C" void kernel_launch(void* const* d_in, const int* in_sizes, int n_in,
                              void* d_out, int out_size, void* d_ws, size_t ws_size,
                              hipStream_t stream) {
  const float* x     = (const float*)d_in[0];  // [4,1024,512]
  const float* wgate = (const float*)d_in[1];  // [512,6]
  const float* gamma = (const float*)d_in[2];  // [512]
  const float* w1    = (const float*)d_in[3];  // [6,512,2048]
  const float* w2    = (const float*)d_in[4];  // [6,2048,512]
  float* out = (float*)d_out;                  // [4,1024,512]

  char* ws = (char*)d_ws;
  size_t off = 0;
  int* counts = (int*)(ws + off);           off += 256;
  int* lists  = (int*)(ws + off);           off += (size_t)NEXP * NTOK * 4;
  float* gatev = (float*)(ws + off);        off += (size_t)NEXP * NTOK * 4;
  _Float16* xnf = (_Float16*)(ws + off);    off += (size_t)NTOK * DMODEL * 2;
  _Float16* w1f = (_Float16*)(ws + off);    off += (size_t)NEXP * DMODEL * HDIM * 2;
  _Float16* w2f = (_Float16*)(ws + off);    off += (size_t)NEXP * HDIM * DMODEL * 2;

  hipMemsetAsync(counts, 0, NEXP * sizeof(int), stream);

  // pack W1 (K=DMODEL, N=HDIM) and W2 (K=HDIM, N=DMODEL) to f16 fragments
  {
    int nfrag = NEXP * (HDIM / 16) * (DMODEL / 32);        // 12288
    int blocks = (nfrag * 32 + 255) / 256;
    pack_w<<<blocks, 256, 0, stream>>>(w1, w1f, DMODEL, HDIM, DMODEL / 32, HDIM / 16);
  }
  {
    int nfrag = NEXP * (DMODEL / 16) * (HDIM / 32);        // 12288
    int blocks = (nfrag * 32 + 255) / 256;
    pack_w<<<blocks, 256, 0, stream>>>(w2, w2f, HDIM, DMODEL, HDIM / 32, DMODEL / 16);
  }

  gate_ln<<<NTOK / 8, 256, 0, stream>>>(x, wgate, gamma, xnf, counts, lists, gatev);

  moe_mlp<<<dim3(NTILES, NEXP), 256, 0, stream>>>(xnf, w1f, w2f, counts, lists,
                                                  gatev, out);
}
